// MultiHeadedAttention_2439541424520
// MI455X (gfx1250) — matrix operand order repaired and checked by reading, not compiled, not run on hardware
//
#include <hip/hip_runtime.h>
#include <hip/hip_bf16.h>
#include <math.h>

// Problem constants (from the reference)
#define BB 2
#define SS 2048
#define DD 1024
#define HH 16
#define DHH 64
#define BS (BB * SS) // 4096 total rows

typedef __bf16 bf16x16 __attribute__((ext_vector_type(16)));
typedef float  floatx8 __attribute__((ext_vector_type(8)));
typedef float  floatx4 __attribute__((ext_vector_type(4)));
typedef unsigned short u16;
typedef u16 u16x8  __attribute__((ext_vector_type(8)));
typedef u16 u16x16 __attribute__((ext_vector_type(16)));
typedef unsigned int uint32x4 __attribute__((ext_vector_type(4)));
typedef int int32x4 __attribute__((ext_vector_type(4)));
typedef int int32x8 __attribute__((ext_vector_type(8)));

// Hardware f32 -> bf16 conversion (let clang pick the cvt instruction)
__device__ __forceinline__ u16 f2bf(float f) {
  return __builtin_bit_cast(u16, (__bf16)f);
}

// A-fragment (16x32 bf16): lane holds two contiguous groups of 8 K-values
__device__ __forceinline__ bf16x16 load_a_bf16(const u16* p0, const u16* p1) {
  u16x8 a0 = *(const u16x8*)p0;
  u16x8 a1 = *(const u16x8*)p1;
  u16x16 t;
#pragma unroll
  for (int i = 0; i < 8; ++i) { t[i] = a0[i]; t[i + 8] = a1[i]; }
  return __builtin_bit_cast(bf16x16, t);
}

// B-fragment (32x16 bf16): lane = column, 16 contiguous K values
__device__ __forceinline__ bf16x16 load_b_contig(const u16* p) {
  u16x16 t = *(const u16x16*)p;
  return __builtin_bit_cast(bf16x16, t);
}

#define WMMA_BF16(A, Bm, C) \
  __builtin_amdgcn_wmma_f32_16x16x32_bf16(false, (A), false, (Bm), (short)0, (C), false, false)

// =====================================================================
// Bulk f32 -> bf16 conversion (bandwidth bound; done ONCE per tensor).
// Sources are dead after this -> nontemporal loads.
// =====================================================================
__global__ __launch_bounds__(256) void cvt_f32_to_bf16(
    const float* __restrict__ src, u16* __restrict__ dst, int n) {
  const int i = (blockIdx.x * 256 + threadIdx.x) * 8;
  if (i + 8 <= n) {
    floatx8 v = __builtin_nontemporal_load((const floatx8*)(src + i));
    u16x8 o;
#pragma unroll
    for (int j = 0; j < 8; ++j) o[j] = f2bf(v[j]);
    *(u16x8*)(dst + i) = o;
  }
}

// =====================================================================
// GEMM: Y[M,N] = X(bf16)[M,K] @ W(bf16)[N,K]^T + bias(f32)
// Block = 256 threads = 8 waves (4 in M x 2 in N), block tile 128x128,
// per-wave tile 32x64 -> 8 WMMA per K-step vs 6 vector loads.
// MODE 0: bf16 out, row-major [M,N]
// MODE 1: bf16 out, V-transposed layout (B,H,DH,S)  (for the V matrix)
// MODE 2: f32 out, row-major [M,N], nontemporal      (final projection)
// =====================================================================
template <int MODE>
__global__ __launch_bounds__(256) void gemm_bf16(
    const u16* __restrict__ X, const u16* __restrict__ W,
    const float* __restrict__ bias, void* __restrict__ Yv,
    int M, int N, int K) {
  const int tid = threadIdx.x;
  const int wave = tid >> 5, lane = tid & 31;
  const int hi = lane >> 4, lo = lane & 15;
  const int wm = wave & 3, wn = wave >> 2;
  const int m0 = blockIdx.x * 128 + wm * 32;   // 2 M-tiles of 16
  const int n0 = blockIdx.y * 128 + wn * 64;   // 4 N-tiles of 16

  floatx8 acc[2][4];
#pragma unroll
  for (int mt = 0; mt < 2; ++mt)
#pragma unroll
    for (int t = 0; t < 4; ++t)
#pragma unroll
      for (int j = 0; j < 8; ++j) acc[mt][t][j] = 0.0f;

  const u16* xrow0 = X + (size_t)(m0 + lo) * K;
  const u16* xrow1 = X + (size_t)(m0 + 16 + lo) * K;
  for (int kb = 0; kb < K; kb += 32) {
    bf16x16 a0 = load_a_bf16(xrow0 + kb + 16 * hi, xrow0 + kb + 8 + 16 * hi);
    bf16x16 a1 = load_a_bf16(xrow1 + kb + 16 * hi, xrow1 + kb + 8 + 16 * hi);
#pragma unroll
    for (int t = 0; t < 4; ++t) {
      const u16* wrow = W + (size_t)(n0 + t * 16 + lo) * K + kb + 16 * hi;
      bf16x16 b = load_b_contig(wrow);
      acc[0][t] = WMMA_BF16(a0, b, acc[0][t]);
      acc[1][t] = WMMA_BF16(a1, b, acc[1][t]);
    }
  }

#pragma unroll
  for (int mt = 0; mt < 2; ++mt) {
#pragma unroll
    for (int t = 0; t < 4; ++t) {
      const int col = n0 + t * 16 + lo;
      const float bv = bias[col];
#pragma unroll
      for (int r = 0; r < 8; ++r) {
        const int row = m0 + mt * 16 + r + 8 * hi;
        const float val = acc[mt][t][r] + bv;
        if (MODE == 0) {
          ((u16*)Yv)[(size_t)row * N + col] = f2bf(val);
        } else if (MODE == 1) {
          // V-transposed: (B,H,DH,S) so PV B-fragments are contiguous
          const int b = row >> 11, s = row & (SS - 1);   // SS = 2048
          const int h = col >> 6, dh = col & (DHH - 1);  // DHH = 64
          ((u16*)Yv)[(((size_t)b * HH + h) * DHH + dh) * SS + s] = f2bf(val);
        } else {
          __builtin_nontemporal_store(val, &((float*)Yv)[(size_t)row * N + col]);
        }
      }
    }
  }
}

// =====================================================================
// Attention core. One block per (b, h, 16-query tile).
// Full 16x2048 f32 score strip + bf16 prob strip live in 320KB LDS.
// LDS layout (dynamic):
//   off      0  float sc[16*2048]    131072 B  raw scores (f32)
//   off 131072  u16   pt[16*2048]     65536 B  probabilities (bf16)
//   off 196608  u16   qt[16*64]        2048 B  Q tile (bf16, TDM target)
//   off 198656  float red[2*256]       2048 B  softmax reductions
//   off 200704  float opart[8*1024]   32768 B  per-wave PV partials
//   off 233472  float mkl[2048]        8192 B  mask row (staged)
// =====================================================================
#define QT_LDS_OFF   196608
#define ATTN_SMEM_BYTES (233472 + 8192)

__global__ __launch_bounds__(256) void attn_core(
    const u16* __restrict__ Qb, const u16* __restrict__ Kb,
    const u16* __restrict__ Vt, const float* __restrict__ mask,
    u16* __restrict__ X2) {
  extern __shared__ char smem[];
  float* sc    = (float*)smem;                  // 16 x 2048 f32
  u16*   pt    = (u16*)(smem + 131072);         // 16 x 2048 bf16
  u16*   qt    = (u16*)(smem + QT_LDS_OFF);     // 16 x 64
  float* red   = (float*)(smem + 198656);       // 2 x 256
  float* opart = (float*)(smem + 200704);       // 8 x 1024
  float* mkl   = (float*)(smem + 233472);       // 2048 mask row

  const int tid  = threadIdx.x;
  const int wave = tid >> 5, lane = tid & 31;
  const int hi = lane >> 4, lo = lane & 15;

  const int qtile = blockIdx.x & 127;       // S/16 = 128
  const int bh    = blockIdx.x >> 7;        // 0..31
  const int b     = bh >> 4;                // H = 16
  const int h     = bh & 15;
  const int q0    = qtile * 16;
  const float* mk = mask + (size_t)b * SS;

  // ---- stage mask row into LDS (vector loads) ----
  for (int i = tid * 4; i < SS; i += 256 * 4)
    *(floatx4*)(mkl + i) = *(const floatx4*)(mk + i);

  // ---- stage Q tile into LDS via Tensor Data Mover ----
#if __has_builtin(__builtin_amdgcn_tensor_load_to_lds)
  if (wave == 0) {
    const unsigned long long ga =
        (unsigned long long)(const void*)(Qb + (size_t)(b * SS + q0) * DD + h * DHH);
    // D# group 0: count=1, lds_addr, 57-bit global addr, type=2
    uint32x4 g0;
    g0[0] = 1u;                                  // count=1, user mode
    g0[1] = (unsigned)QT_LDS_OFF;                // lds_addr (bytes)
    g0[2] = (unsigned)(ga & 0xFFFFFFFFu);        // global_addr[31:0]
    g0[3] = (unsigned)((ga >> 32) & 0x01FFFFFFu) | (2u << 30); // addr[56:32] | type=2
    // D# group 1: data_size=2B, tensor dims (large), tile 64x16, stride=DD
    const unsigned TD = 1u << 20;                // generous tensor extent
    int32x8 g1;
    g1[0] = (int)(1u << 16);                     // data_size = 1 -> 2 bytes
    g1[1] = (int)((TD & 0xFFFFu) << 16);         // tensor_dim0[15:0] @ bits 63:48
    g1[2] = (int)(((TD >> 16) & 0xFFFFu) | ((TD & 0xFFFFu) << 16)); // dim0 hi | dim1 lo
    g1[3] = (int)(((TD >> 16) & 0xFFFFu) | (64u << 16));  // dim1 hi | tile_dim0=64
    g1[4] = (int)16u;                            // tile_dim1 = 16 rows
    g1[5] = (int)DD;                             // tensor_dim0_stride[31:0] = 1024
    g1[6] = 0;                                   // stride0 hi | stride1 lo
    g1[7] = 0;
    int32x4 gz4; gz4[0] = 0; gz4[1] = 0; gz4[2] = 0; gz4[3] = 0;
    int32x8 gz8;
#pragma unroll
    for (int j = 0; j < 8; ++j) gz8[j] = 0;
    // 6-arg form (this toolchain): (g0, g1, g2, g3, g4, cpol)
    __builtin_amdgcn_tensor_load_to_lds(g0, g1, gz4, gz4, gz8, 0);
    __builtin_amdgcn_s_wait_tensorcnt(0);
  }
#else
  for (int i = tid; i < 16 * DHH; i += 256) {
    const int r = i >> 6, c = i & 63;
    qt[i] = Qb[(size_t)(b * SS + q0 + r) * DD + h * DHH + c];
  }
#endif
  __syncthreads();

  // ---- scores = Q K^T / 8 ----
  {
    // Q A-fragments are invariant across all 128 score tiles: hoist.
    const bf16x16 aq0 = load_a_bf16(qt + lo * DHH + 16 * hi,
                                    qt + lo * DHH + 8 + 16 * hi);
    const bf16x16 aq1 = load_a_bf16(qt + lo * DHH + 32 + 16 * hi,
                                    qt + lo * DHH + 40 + 16 * hi);
    // K row pointer for this wave's first tile; advance by 128 rows/iter.
    const u16* kptr = Kb + (size_t)(b * SS + wave * 16 + lo) * DD + h * DHH;
    const size_t kstep = (size_t)128 * DD;   // 8 waves x 16 keys
    float* scp = sc + (8 * hi) * SS + wave * 16 + lo;
    for (int ct = wave; ct < SS / 16; ct += 8) {
      __builtin_prefetch(kptr + kstep, 0, 1);   // next tile, unconditional
      floatx8 acc;
      bf16x16 bm0 = load_b_contig(kptr + 16 * hi);
      bf16x16 bm1 = load_b_contig(kptr + 32 + 16 * hi);
      acc = __builtin_amdgcn_wmma_f32_16x16x32_bf16(
          false, aq0, false, bm0, (short)0, (floatx8){0,0,0,0,0,0,0,0}, false, false);
      acc = WMMA_BF16(aq1, bm1, acc);
#pragma unroll
      for (int r = 0; r < 8; ++r) scp[r * SS] = acc[r] * 0.125f;
      kptr += kstep;
      scp  += 128;
    }
  }
  __syncthreads();

  // ---- masked softmax (AllenNLP semantics); emit bf16 probs ----
  {
    const int row = tid >> 4, t = tid & 15;
    float* srow = sc + (size_t)row * SS;
    u16*   prow = pt + (size_t)row * SS;
    const float mq = mkl[q0 + row];

    float lmax = -3.0e38f;
    for (int c = t; c < SS; c += 16) {
      const float v = srow[c] * (mq * mkl[c]);  // s * m
      srow[c] = v;
      lmax = fmaxf(lmax, v);
    }
    red[row * 16 + t] = lmax;
    __syncthreads();
    float rmax = -3.0e38f;
#pragma unroll
    for (int i = 0; i < 16; ++i) rmax = fmaxf(rmax, red[row * 16 + i]);
    __syncthreads();

    float se = 0.0f, sem = 0.0f;
    for (int c = t; c < SS; c += 16) {
      const float e  = __expf(srow[c] - rmax);
      const float em = e * (mq * mkl[c]);       // softmax numerator * m
      srow[c] = em;
      se += e; sem += em;
    }
    red[row * 16 + t]       = se;
    red[256 + row * 16 + t] = sem;
    __syncthreads();
    float tse = 0.0f, tsem = 0.0f;
#pragma unroll
    for (int i = 0; i < 16; ++i) { tse += red[row * 16 + i]; tsem += red[256 + row * 16 + i]; }
    // p = (em/se) / (sem/se + 1e-13) = em / (sem + se*1e-13)
    const float scale = 1.0f / (tsem + tse * 1e-13f);
    for (int c = t; c < SS; c += 16) prow[c] = f2bf(srow[c] * scale);
  }
  __syncthreads();

  // ---- O_partial = P(chunk) @ V(chunk), each wave takes 256 keys ----
  {
    floatx8 acc[4];
#pragma unroll
    for (int t = 0; t < 4; ++t)
#pragma unroll
      for (int j = 0; j < 8; ++j) acc[t][j] = 0.0f;

    const int k0 = wave * 256;
    const u16* vbase = Vt + ((size_t)b * HH + h) * DHH * SS;
    for (int kb = 0; kb < 256; kb += 32) {
      const int kk = k0 + kb;
      bf16x16 a = load_a_bf16(pt + (size_t)lo * SS + kk + 16 * hi,
                              pt + (size_t)lo * SS + kk + 8 + 16 * hi);
#pragma unroll
      for (int t = 0; t < 4; ++t) {
        // B[k][n] = Vt[(t*16+n)][kk+k]  -> contiguous row read of Vt
        const u16* vrow = vbase + (size_t)(t * 16 + lo) * SS + kk + 16 * hi;
        bf16x16 bm = load_b_contig(vrow);
        acc[t] = WMMA_BF16(a, bm, acc[t]);
      }
    }
    float* op = opart + wave * 1024;
#pragma unroll
    for (int t = 0; t < 4; ++t)
#pragma unroll
      for (int r = 0; r < 8; ++r)
        op[(r + 8 * hi) * DHH + t * 16 + lo] = acc[t][r];
  }
  __syncthreads();

  // ---- reduce 8 wave partials, emit bf16 in (B,S,D) layout ----
  for (int i = tid; i < 16 * DHH; i += 256) {
    float s = 0.0f;
#pragma unroll
    for (int w = 0; w < 8; ++w) s += opart[w * 1024 + i];
    const int r = i >> 6, c = i & 63;
    X2[(size_t)(b * SS + q0 + r) * DD + h * DHH + c] = f2bf(s);
  }
}

// =====================================================================
// Host launch
// =====================================================================
extern "C" void kernel_launch(void* const* d_in, const int* in_sizes, int n_in,
                              void* d_out, int out_size, void* d_ws, size_t ws_size,
                              hipStream_t stream) {
  (void)in_sizes; (void)n_in; (void)out_size; (void)ws_size;
  const float* inputs = (const float*)d_in[0];
  const float* maskp  = (const float*)d_in[1];
  const float* Wq = (const float*)d_in[2];
  const float* bq = (const float*)d_in[3];
  const float* Wk = (const float*)d_in[4];
  const float* bk = (const float*)d_in[5];
  const float* Wv = (const float*)d_in[6];
  const float* bv = (const float*)d_in[7];
  const float* Wo = (const float*)d_in[8];
  const float* bo = (const float*)d_in[9];
  float* out = (float*)d_out;

  // Workspace layout (u16 units)
  const size_t nX = (size_t)BS * DD;   // 4M activations
  const size_t nW = (size_t)DD * DD;   // 1M per weight
  u16* Xb  = (u16*)d_ws;               // inputs, bf16
  u16* Wqb = Xb  + nX;
  u16* Wkb = Wqb + nW;
  u16* Wvb = Wkb + nW;
  u16* Wob = Wvb + nW;
  u16* Qb  = Wob + nW;                 // Q (B,S,D) bf16
  u16* Kb  = Qb  + nX;                 // K (B,S,D) bf16
  u16* Vt  = Kb  + nX;                 // V (B,H,DH,S) bf16 (transposed)
  u16* X2  = Vt  + nX;                 // attention output (B,S,D) bf16

  // One-time f32 -> bf16 conversions (bandwidth bound)
  cvt_f32_to_bf16<<<(int)(nX / (8 * 256)), 256, 0, stream>>>(inputs, Xb, (int)nX);
  cvt_f32_to_bf16<<<(int)(nW / (8 * 256)), 256, 0, stream>>>(Wq, Wqb, (int)nW);
  cvt_f32_to_bf16<<<(int)(nW / (8 * 256)), 256, 0, stream>>>(Wk, Wkb, (int)nW);
  cvt_f32_to_bf16<<<(int)(nW / (8 * 256)), 256, 0, stream>>>(Wv, Wvb, (int)nW);
  cvt_f32_to_bf16<<<(int)(nW / (8 * 256)), 256, 0, stream>>>(Wo, Wob, (int)nW);

  dim3 ggrid(BS / 128, DD / 128);      // (32, 8)
  gemm_bf16<0><<<ggrid, 256, 0, stream>>>(Xb, Wqb, bq, (void*)Qb, BS, DD, DD);
  gemm_bf16<0><<<ggrid, 256, 0, stream>>>(Xb, Wkb, bk, (void*)Kb, BS, DD, DD);
  gemm_bf16<1><<<ggrid, 256, 0, stream>>>(Xb, Wvb, bv, (void*)Vt, BS, DD, DD);

  (void)hipFuncSetAttribute((const void*)attn_core,
                            hipFuncAttributeMaxDynamicSharedMemorySize,
                            ATTN_SMEM_BYTES);
  attn_core<<<BB * HH * (SS / 16), 256, ATTN_SMEM_BYTES, stream>>>(Qb, Kb, Vt, maskp, X2);

  gemm_bf16<2><<<ggrid, 256, 0, stream>>>(X2, Wob, bo, (void*)out, BS, DD, DD);
}
